// RSSM_54692113547773
// MI455X (gfx1250) — matrix operand order repaired and checked by reading, not compiled, not run on hardware
//
#include <hip/hip_runtime.h>
#include <hip/hip_bf16.h>
#include <math.h>

#ifndef __has_builtin
#define __has_builtin(x) 0
#endif

// ---------------------------------------------------------------------------
// RSSM obs_step for MI455X (gfx1250, wave32, WMMA bf16 16x16x32,
// async global->LDS double-buffered pipeline)
// ---------------------------------------------------------------------------

typedef __bf16 bf16_t;
typedef __attribute__((ext_vector_type(16))) __bf16 v16bf;
typedef __attribute__((ext_vector_type(8)))  __bf16 v8bf;
typedef __attribute__((ext_vector_type(8)))  float  v8f;

#define BS   16384
#define SDIM 1024
#define ADIM 64
#define HID  1024
#define REC  1024
#define LAT  1024
#define K1   (SDIM + ADIM)   // 1088
#define K2   2048            // [x, deter] / [h, obs]
#define N2   3072            // 3*REC

#define TILE_M 128
#define GT_N   128           // generic gemm N tile
#define GR_N   64            // gru gemm N tile
#define KSTEP  32
#define LDS_STRIDE 48        // bf16 elements per LDS row (96B, 16B-aligned chunks)

// ---------------------------------------------------------------------------
// CDNA5 async global->LDS copies (ASYNCcnt).  Signatures per clang-22
// diagnostics: (addrspace(1) int4*, addrspace(3) int4*, imm offset, imm cpol)
// ---------------------------------------------------------------------------
typedef int v4i __attribute__((ext_vector_type(4)));
typedef int v2i __attribute__((ext_vector_type(2)));
typedef __attribute__((address_space(1))) v4i g_v4i;
typedef __attribute__((address_space(3))) v4i l_v4i;
typedef __attribute__((address_space(1))) v2i g_v2i;
typedef __attribute__((address_space(3))) v2i l_v2i;

__device__ __forceinline__ void async_copy_b128(const void* g, void* l) {
#if __has_builtin(__builtin_amdgcn_global_load_async_to_lds_b128)
    __builtin_amdgcn_global_load_async_to_lds_b128((g_v4i*)g, (l_v4i*)l, 0, 0);
#else
    unsigned loff = (unsigned)(unsigned long long)(__attribute__((address_space(3))) char*)l;
    asm volatile("global_load_async_to_lds_b128 %0, %1, off"
                 :: "v"(loff), "v"((unsigned long long)(size_t)g) : "memory");
#endif
}
__device__ __forceinline__ void async_copy_b64(const void* g, void* l) {
#if __has_builtin(__builtin_amdgcn_global_load_async_to_lds_b64)
    __builtin_amdgcn_global_load_async_to_lds_b64((g_v2i*)g, (l_v2i*)l, 0, 0);
#else
    unsigned loff = (unsigned)(unsigned long long)(__attribute__((address_space(3))) char*)l;
    asm volatile("global_load_async_to_lds_b64 %0, %1, off"
                 :: "v"(loff), "v"((unsigned long long)(size_t)g) : "memory");
#endif
}
#if __has_builtin(__builtin_amdgcn_s_wait_asynccnt)
#define WAIT_ASYNC(n) __builtin_amdgcn_s_wait_asynccnt(n)
#else
#define WAIT_ASYNC(n) asm volatile("s_wait_asynccnt %0" :: "n"(n) : "memory")
#endif

// ---------------------------------------------------------------------------
// Fragment loaders matching CDNA5 WMMA VGPR layouts (ISA 7.12.2)
// ---------------------------------------------------------------------------
__device__ __forceinline__ v16bf load_frag_a(const bf16_t* As, int row, int khalf) {
    const v8bf lo = *(const v8bf*)(As + row * LDS_STRIDE + khalf * 8);
    const v8bf hi = *(const v8bf*)(As + row * LDS_STRIDE + 16 + khalf * 8);
    return __builtin_shufflevector(lo, hi, 0,1,2,3,4,5,6,7,8,9,10,11,12,13,14,15);
}
__device__ __forceinline__ v16bf load_frag_b(const bf16_t* Bs, int n, int khalf) {
    const v8bf lo = *(const v8bf*)(Bs + n * LDS_STRIDE + khalf * 8);
    const v8bf hi = *(const v8bf*)(Bs + n * LDS_STRIDE + 16 + khalf * 8);
    return __builtin_shufflevector(lo, hi, 0,1,2,3,4,5,6,7,8,9,10,11,12,13,14,15);
}
__device__ __forceinline__ v8f wmma_bf16(v16bf a, v16bf b, v8f c) {
    return __builtin_amdgcn_wmma_f32_16x16x32_bf16(false, a, false, b, (short)0, c, false, false);
}
__device__ __forceinline__ float sigmoidf_(float x) { return 1.0f / (1.0f + __expf(-x)); }

// ---------------------------------------------------------------------------
// Generic bf16 GEMM:  out[m][n] = sum_k A[m][k] * W[n][k] + bias[n]
// Block tile 128x128, 512 threads = 16 waves (4M x 4N), wave tile 32x32.
// Double-buffered async global->LDS pipeline.
// ---------------------------------------------------------------------------
__global__ __launch_bounds__(512)
void gemm_bf16_kernel(const bf16_t* __restrict__ A, int lda,
                      const bf16_t* __restrict__ W, int ldw,
                      const float* __restrict__ bias, int nk,
                      float* __restrict__ outF, bf16_t* __restrict__ outB,
                      int ld_out) {
    __shared__ alignas(16) bf16_t As[2][TILE_M * LDS_STRIDE];
    __shared__ alignas(16) bf16_t Bs[2][GT_N * LDS_STRIDE];

    const int tid   = threadIdx.x;
    const int mbase = blockIdx.x * TILE_M;
    const int nbase = blockIdx.y * GT_N;
    const int w     = tid >> 5, lane = tid & 31;
    const int wm    = w & 3,   wn   = w >> 2;
    const int khalf = lane >> 4, lcol = lane & 15;

    const int arow = tid >> 2, acol = (tid & 3) * 8;   // 16B per thread, 128x32 tile

    v8f acc00 = {}, acc01 = {}, acc10 = {}, acc11 = {};

    // stage 0
    async_copy_b128(&A[(size_t)(mbase + arow) * lda + acol], &As[0][arow * LDS_STRIDE + acol]);
    async_copy_b128(&W[(size_t)(nbase + arow) * ldw + acol], &Bs[0][arow * LDS_STRIDE + acol]);

    for (int kt = 0; kt < nk; ++kt) {
        const int cur = kt & 1;
        if (kt + 1 < nk) {
            const int kb = (kt + 1) * KSTEP;
            async_copy_b128(&A[(size_t)(mbase + arow) * lda + kb + acol],
                            &As[cur ^ 1][arow * LDS_STRIDE + acol]);
            async_copy_b128(&W[(size_t)(nbase + arow) * ldw + kb + acol],
                            &Bs[cur ^ 1][arow * LDS_STRIDE + acol]);
            WAIT_ASYNC(2);         // current stage complete (in-order)
        } else {
            WAIT_ASYNC(0);
        }
        __syncthreads();

        v16bf a0 = load_frag_a(As[cur], wm * 32 + lcol, khalf);
        v16bf a1 = load_frag_a(As[cur], wm * 32 + 16 + lcol, khalf);
        v16bf b0 = load_frag_b(Bs[cur], wn * 32 + lcol, khalf);
        v16bf b1 = load_frag_b(Bs[cur], wn * 32 + 16 + lcol, khalf);
        acc00 = wmma_bf16(a0, b0, acc00);
        acc01 = wmma_bf16(a0, b1, acc01);
        acc10 = wmma_bf16(a1, b0, acc10);
        acc11 = wmma_bf16(a1, b1, acc11);
        __syncthreads();
    }

    const int gc0 = nbase + wn * 32 + lcol;
    const int gc1 = gc0 + 16;
    const float bv0 = bias[gc0];
    const float bv1 = bias[gc1];
#pragma unroll
    for (int e = 0; e < 8; ++e) {
        const int gr0 = mbase + wm * 32 + khalf * 8 + e;
        const int gr1 = gr0 + 16;
        const float v00 = acc00[e] + bv0, v01 = acc01[e] + bv1;
        const float v10 = acc10[e] + bv0, v11 = acc11[e] + bv1;
        if (outB) {
            outB[(size_t)gr0 * ld_out + gc0] = (bf16_t)v00;
            outB[(size_t)gr0 * ld_out + gc1] = (bf16_t)v01;
            outB[(size_t)gr1 * ld_out + gc0] = (bf16_t)v10;
            outB[(size_t)gr1 * ld_out + gc1] = (bf16_t)v11;
        } else {
            outF[(size_t)gr0 * ld_out + gc0] = v00;
            outF[(size_t)gr0 * ld_out + gc1] = v01;
            outF[(size_t)gr1 * ld_out + gc0] = v10;
            outF[(size_t)gr1 * ld_out + gc1] = v11;
        }
    }
}

// ---------------------------------------------------------------------------
// Fused GRU GEMM.  A2 = [x, deter] (BSx2048 bf16), Wcat[n][k] =
//   k<1024 ? W_ih[n][k] : W_hh[n][k-1024]   (3072x2048 bf16).
// Accumulator sets: R/Z over full K; N (i_n) over K[0,1024);
// Hn (h_n) over K[1024,2048).  Epilogue applies GRU nonlinearity.
// ---------------------------------------------------------------------------
__global__ __launch_bounds__(512)
void gru_gemm_kernel(const bf16_t* __restrict__ A2,
                     const bf16_t* __restrict__ Wcat,
                     const float* __restrict__ bias2,
                     const float* __restrict__ b_ih,
                     const float* __restrict__ b_hh,
                     const float* __restrict__ deter,
                     const unsigned char* __restrict__ first,
                     float* __restrict__ out_h,
                     bf16_t* __restrict__ A4) {
    __shared__ alignas(16) bf16_t As[2][TILE_M * LDS_STRIDE];
    __shared__ alignas(16) bf16_t Bs[2][3][GR_N * LDS_STRIDE];

    const int tid   = threadIdx.x;
    const int mbase = blockIdx.x * TILE_M;
    const int nbase = blockIdx.y * GR_N;
    const int w     = tid >> 5, lane = tid & 31;
    const int wm    = w & 3,   wn   = w >> 2;
    const int khalf = lane >> 4, lcol = lane & 15;

    const int arow = tid >> 2, acol = (tid & 3) * 8;   // A: 128x32, 16B/thread
    const int brow = tid >> 3, bcol = (tid & 7) * 4;   // B: 64x32,  8B/thread

    v8f aR0 = {}, aR1 = {}, aZ0 = {}, aZ1 = {};
    v8f aN0 = {}, aN1 = {}, aH0 = {}, aH1 = {};

    const int NK = K2 / KSTEP;  // 64

    // stage 0
    async_copy_b128(&A2[(size_t)(mbase + arow) * K2 + acol], &As[0][arow * LDS_STRIDE + acol]);
#pragma unroll
    for (int j = 0; j < 3; ++j)
        async_copy_b64(&Wcat[(size_t)(j * REC + nbase + brow) * K2 + bcol],
                       &Bs[0][j][brow * LDS_STRIDE + bcol]);

    for (int kt = 0; kt < NK; ++kt) {
        const int cur = kt & 1;
        if (kt + 1 < NK) {
            const int kb = (kt + 1) * KSTEP;
            async_copy_b128(&A2[(size_t)(mbase + arow) * K2 + kb + acol],
                            &As[cur ^ 1][arow * LDS_STRIDE + acol]);
#pragma unroll
            for (int j = 0; j < 3; ++j)
                async_copy_b64(&Wcat[(size_t)(j * REC + nbase + brow) * K2 + kb + bcol],
                               &Bs[cur ^ 1][j][brow * LDS_STRIDE + bcol]);
            WAIT_ASYNC(4);
        } else {
            WAIT_ASYNC(0);
        }
        __syncthreads();

        v16bf a0 = load_frag_a(As[cur], wm * 32 + lcol, khalf);
        v16bf a1 = load_frag_a(As[cur], wm * 32 + 16 + lcol, khalf);
        v16bf bR = load_frag_b(Bs[cur][0], wn * 16 + lcol, khalf);
        v16bf bZ = load_frag_b(Bs[cur][1], wn * 16 + lcol, khalf);
        v16bf bN = load_frag_b(Bs[cur][2], wn * 16 + lcol, khalf);
        aR0 = wmma_bf16(a0, bR, aR0);  aR1 = wmma_bf16(a1, bR, aR1);
        aZ0 = wmma_bf16(a0, bZ, aZ0);  aZ1 = wmma_bf16(a1, bZ, aZ1);
        if (kt < NK / 2) {              // k in [0,1024): i_n contribution
            aN0 = wmma_bf16(a0, bN, aN0);  aN1 = wmma_bf16(a1, bN, aN1);
        } else {                        // k in [1024,2048): h_n contribution
            aH0 = wmma_bf16(a0, bN, aH0);  aH1 = wmma_bf16(a1, bN, aH1);
        }
        __syncthreads();
    }

    const int gc = nbase + wn * 16 + lcol;
    const float br  = bias2[gc];
    const float bz  = bias2[REC + gc];
    const float bin = b_ih[2 * REC + gc];
    const float bhn = b_hh[2 * REC + gc];
#pragma unroll
    for (int e = 0; e < 8; ++e) {
        const int gr0 = mbase + wm * 32 + khalf * 8 + e;
        const int gr1 = gr0 + 16;
        {
            const float r = sigmoidf_(aR0[e] + br);
            const float z = sigmoidf_(aZ0[e] + bz);
            const float n = tanhf(aN0[e] + bin + r * (aH0[e] + bhn));
            const float d = first[gr0] ? 0.0f : deter[(size_t)gr0 * REC + gc];
            const float h = (1.0f - z) * n + z * d;
            out_h[(size_t)gr0 * REC + gc] = h;
            A4[(size_t)gr0 * K2 + gc]     = (bf16_t)h;
        }
        {
            const float r = sigmoidf_(aR1[e] + br);
            const float z = sigmoidf_(aZ1[e] + bz);
            const float n = tanhf(aN1[e] + bin + r * (aH1[e] + bhn));
            const float d = first[gr1] ? 0.0f : deter[(size_t)gr1 * REC + gc];
            const float h = (1.0f - z) * n + z * d;
            out_h[(size_t)gr1 * REC + gc] = h;
            A4[(size_t)gr1 * K2 + gc]     = (bf16_t)h;
        }
    }
}

// ---------------------------------------------------------------------------
// Elementwise prep kernels
// ---------------------------------------------------------------------------
__global__ void f32_to_bf16_kernel(const float* __restrict__ src,
                                   bf16_t* __restrict__ dst, int n) {
    int i = blockIdx.x * blockDim.x + threadIdx.x;
    if (i < n) dst[i] = (bf16_t)src[i];
}

__global__ void build_wcat_kernel(const float* __restrict__ W_ih,
                                  const float* __restrict__ W_hh,
                                  bf16_t* __restrict__ Wcat, int n) {
    int i = blockIdx.x * blockDim.x + threadIdx.x;
    if (i >= n) return;
    int row = i >> 11;            // / 2048
    int k   = i & 2047;
    float v = (k < REC) ? W_ih[(size_t)row * REC + k]
                        : W_hh[(size_t)row * REC + (k - REC)];
    Wcat[i] = (bf16_t)v;
}

__global__ void build_bias2_kernel(const float* __restrict__ b_ih,
                                   const float* __restrict__ b_hh,
                                   float* __restrict__ bias2, int n) {
    int i = blockIdx.x * blockDim.x + threadIdx.x;
    if (i < n) bias2[i] = b_ih[i] + b_hh[i];
}

__global__ void build_a1_kernel(const float* __restrict__ stoch,
                                const float* __restrict__ action,
                                const unsigned char* __restrict__ first,
                                bf16_t* __restrict__ A1, int n) {
    int i = blockIdx.x * blockDim.x + threadIdx.x;
    if (i >= n) return;
    int b = i / K1;
    int k = i - b * K1;
    float v;
    if (k < SDIM) v = first[b] ? 0.0f : stoch[(size_t)b * SDIM + k];
    else          v = action[(size_t)b * ADIM + (k - SDIM)];
    A1[i] = (bf16_t)v;
}

__global__ void build_second_half_kernel(const float* __restrict__ src,
                                         const unsigned char* __restrict__ first,
                                         bf16_t* __restrict__ dst, int n,
                                         int use_first) {
    int i = blockIdx.x * blockDim.x + threadIdx.x;
    if (i >= n) return;
    int b = i >> 10;              // / 1024
    int k = i & 1023;
    float v = src[i];
    if (use_first && first[b]) v = 0.0f;
    dst[(size_t)b * K2 + REC + k] = (bf16_t)v;
}

// ---------------------------------------------------------------------------
// Gumbel one-hot sampling: one wave32 per (b, s) group, D=32 lanes.
// ---------------------------------------------------------------------------
__device__ __forceinline__ unsigned int hash_u32(unsigned int x) {
    x ^= x >> 16; x *= 0x7feb352dU;
    x ^= x >> 15; x *= 0x846ca68bU;
    x ^= x >> 16; return x;
}

__global__ void sample_onehot_kernel(const float* __restrict__ logits,
                                     float* __restrict__ onehot) {
    const int gid  = blockIdx.x * (blockDim.x >> 5) + (threadIdx.x >> 5);
    const int lane = threadIdx.x & 31;
    const int b = gid >> 5;
    const int s = gid & 31;
    const size_t off = (size_t)b * LAT + s * 32 + lane;

    const float l = logits[off];
    const unsigned int idx = (unsigned int)(gid * 32 + lane);
    const unsigned int u32 = hash_u32(idx * 2654435761u ^ hash_u32(idx ^ 42u));
    const float u = (float)(u32 >> 8) * (1.0f / 16777216.0f) + 1e-7f;
    const float g = -__logf(-__logf(u));

    float val = l + g;
    int   best = lane;
#pragma unroll
    for (int offm = 16; offm > 0; offm >>= 1) {
        float ov = __shfl_xor(val, offm, 32);
        int   oi = __shfl_xor(best, offm, 32);
        if (ov > val || (ov == val && oi < best)) { val = ov; best = oi; }
    }
    onehot[off] = (lane == best) ? 1.0f : 0.0f;
}

// ---------------------------------------------------------------------------
// Host launcher
// ---------------------------------------------------------------------------
static inline int cdiv(int a, int b) { return (a + b - 1) / b; }

extern "C" void kernel_launch(void* const* d_in, const int* in_sizes, int n_in,
                              void* d_out, int out_size, void* d_ws, size_t ws_size,
                              hipStream_t stream) {
    const float* stoch  = (const float*)d_in[0];
    const float* deter  = (const float*)d_in[1];
    const float* action = (const float*)d_in[3];
    const float* obs    = (const float*)d_in[4];
    const unsigned char* first = (const unsigned char*)d_in[5];
    const float* W_in   = (const float*)d_in[6];
    const float* b_in   = (const float*)d_in[7];
    const float* W_ih   = (const float*)d_in[8];
    const float* b_ih   = (const float*)d_in[9];
    const float* W_hh   = (const float*)d_in[10];
    const float* b_hh   = (const float*)d_in[11];
    const float* W_post = (const float*)d_in[14];
    const float* b_post = (const float*)d_in[15];

    float* out_h      = (float*)d_out;                    // (BS, REC)
    float* out_onehot = out_h + (size_t)BS * REC;         // (BS, LAT)
    float* out_logits = out_onehot + (size_t)BS * LAT;    // (BS, LAT)

    char* ws = (char*)d_ws;
    size_t off = 0;
    bf16_t* A2    = (bf16_t*)(ws + off); off += (size_t)BS * K2 * 2;      // [x, deter]
    bf16_t* A4    = (bf16_t*)(ws + off); off += (size_t)BS * K2 * 2;      // [h, obs]
    bf16_t* A1    = (bf16_t*)(ws + off); off += (size_t)BS * K1 * 2;      // [stoch, action]
    bf16_t* Wb_in = (bf16_t*)(ws + off); off += (size_t)HID * K1 * 2;
    bf16_t* Wcat  = (bf16_t*)(ws + off); off += (size_t)N2 * K2 * 2;
    bf16_t* Wb_po = (bf16_t*)(ws + off); off += (size_t)LAT * K2 * 2;
    float*  bias2 = (float*)(ws + off);  off += (size_t)N2 * 4;
    (void)ws_size; (void)in_sizes; (void)n_in; (void)out_size;

    const int TPB = 256;

    // --- prep: weights to bf16 ---
    {
        int n = HID * K1;
        f32_to_bf16_kernel<<<cdiv(n, TPB), TPB, 0, stream>>>(W_in, Wb_in, n);
    }
    {
        int n = N2 * K2;
        build_wcat_kernel<<<cdiv(n, TPB), TPB, 0, stream>>>(W_ih, W_hh, Wcat, n);
    }
    {
        int n = LAT * K2;
        f32_to_bf16_kernel<<<cdiv(n, TPB), TPB, 0, stream>>>(W_post, Wb_po, n);
    }
    build_bias2_kernel<<<cdiv(N2, TPB), TPB, 0, stream>>>(b_ih, b_hh, bias2, N2);

    // --- prep: activation panels ---
    {
        int n = BS * K1;
        build_a1_kernel<<<cdiv(n, TPB), TPB, 0, stream>>>(stoch, action, first, A1, n);
    }
    {
        int n = BS * REC;  // deter -> A2[:,1024:2048], masked by first
        build_second_half_kernel<<<cdiv(n, TPB), TPB, 0, stream>>>(deter, first, A2, n, 1);
    }
    {
        int n = BS * HID;  // obs -> A4[:,1024:2048]
        build_second_half_kernel<<<cdiv(n, TPB), TPB, 0, stream>>>(obs, first, A4, n, 0);
    }

    // --- GEMM1: x = A1 @ W_in^T + b_in  -> bf16 into A2[:,0:1024] ---
    {
        dim3 grid(BS / TILE_M, HID / GT_N);
        gemm_bf16_kernel<<<grid, 512, 0, stream>>>(A1, K1, Wb_in, K1, b_in,
                                                   K1 / KSTEP, nullptr, A2, K2);
    }

    // --- fused GRU GEMM: h = GRU(A2) -> out_h (f32) + A4[:,0:1024] (bf16) ---
    {
        dim3 grid(BS / TILE_M, REC / GR_N);
        gru_gemm_kernel<<<grid, 512, 0, stream>>>(A2, Wcat, bias2, b_ih, b_hh,
                                                  deter, first, out_h, A4);
    }

    // --- GEMM4: logits = A4 @ W_post^T + b_post -> f32 out ---
    {
        dim3 grid(BS / TILE_M, LAT / GT_N);
        gemm_bf16_kernel<<<grid, 512, 0, stream>>>(A4, K2, Wb_po, K2, b_post,
                                                   K2 / KSTEP, out_logits, nullptr, LAT);
    }

    // --- sampling: one wave per (b, s) group of 32 categories ---
    {
        int groups = BS * 32;              // 524288 groups
        int blocks = groups / 8;           // 8 waves (256 thr) per block
        sample_onehot_kernel<<<blocks, 256, 0, stream>>>(out_logits, out_onehot);
    }
}
